// CoSemiGNN_54202487275568
// MI455X (gfx1250) — matrix operand order repaired and checked by reading, not compiled, not run on hardware
//
#include <hip/hip_runtime.h>
#include <hip/hip_bf16.h>

// ---------------------------------------------------------------------------
// CDNA5 (gfx1250, wave32) implementation of the CoSemiGNN forward pass.
// Dense GEMMs: double-buffered LDS bf16 tiles -> v_wmma_f32_16x16x32_bf16,
// 4 accumulators per wave (16x64 wave tile, 64x128 block tile).
// Edge segment ops use global float atomics; row ops use wave32 shfl trees.
// ---------------------------------------------------------------------------

typedef __attribute__((ext_vector_type(16))) __bf16 bf16x16;
typedef __attribute__((ext_vector_type(8)))  float  f32x8;

#define GNN_N    32768
#define GNN_E    262144
#define GNN_ETOT (GNN_E + GNN_N)

__device__ __forceinline__ float lrelu(float x, float s) {
  return x >= 0.f ? x : s * x;
}

__device__ __forceinline__ float wave_sum32(float v) {
  #pragma unroll
  for (int o = 16; o > 0; o >>= 1) v += __shfl_xor(v, o, 32);
  return v;
}

// order-preserving float <-> uint encoding for atomic segment-max
__device__ __forceinline__ unsigned enc_f32(float f) {
  unsigned b = __float_as_uint(f);
  return b ^ ((unsigned)((int)b >> 31) | 0x80000000u);
}
__device__ __forceinline__ float dec_f32(unsigned e) {
  unsigned b = e ^ ((unsigned)((int)(~e) >> 31) | 0x80000000u);
  return __uint_as_float(b);
}

union FragU { bf16x16 v; uint4 q[2]; };
union Pack8 { __bf16 h[8]; uint4 q; };

// ---------------------------------------------------------------------------
// WMMA GEMM: C[rows x M] = act( A[rows x K] @ B + bias ),  B = W^T.
//   WT == 0 : W stored [M x K] row-major (standard y = x @ W.T)
//   WT == 1 : W stored [K x M] row-major (y = x @ W)
// Block = 256 threads = 8 waves. Block tile 64x128, K-step 32.
// Wave tile 16x64 = 4 WMMA accumulators sharing one A fragment.
// Ping-pong LDS buffers; next global tile loads issue before current WMMAs.
// rows % 64 == 0, M % 128 == 0, K % 32 == 0 at every call site.
// ---------------------------------------------------------------------------
template <int WT, int ACT>
__global__ __launch_bounds__(256) void gemm_wmma_kernel(
    const float* __restrict__ A, const float* __restrict__ W,
    const float* __restrict__ bias, float* __restrict__ C,
    int K, int lda, int ldw, int ldc)
{
  __shared__ __align__(16) __bf16 lsA[2][64 * 32];    // [buf][row][k]   2x4 KB
  __shared__ __align__(16) __bf16 lsB[2][128 * 32];   // [buf][col][k]   2x8 KB

  const int tid  = threadIdx.x;
  const int lane = tid & 31;
  const int wave = tid >> 5;
  const int rs   = wave >> 1;          // row strip 0..3
  const int ch   = wave & 1;           // col half  0..1 (64 cols each)

  const int rowBase0 = blockIdx.y * 64;
  const int colBase0 = blockIdx.x * 128;

  const int  mrow = lane & 15;
  const bool hi   = lane >= 16;

  // A staging: 64 rows x 32 K, 8 floats/thread
  const int sR = tid >> 2;                  // 0..63
  const int sK = (tid & 3) * 8;             // 0,8,16,24
  const float* gA = A + (size_t)(rowBase0 + sR) * lda + sK;

  // B staging: 128 cols x 32 K, 16 floats/thread
  const float* gB;
  int tK = 0, tM = 0, sR2 = 0, sK2 = 0;
  if (WT == 0) {
    sR2 = tid >> 1;                     // 0..127
    sK2 = (tid & 1) * 16;               // 0,16
    gB = W + (size_t)(colBase0 + sR2) * ldw + sK2;      // W[m][k]
  } else {
    tK = tid >> 3;                      // 0..31
    tM = (tid & 7) * 16;                // 0..112
    gB = W + (size_t)tK * ldw + colBase0 + tM;          // W[k][m]
  }

  // fragment LDS byte offsets (constant across K loop)
  const int aOff = (rs * 16 + mrow) * 64 + (hi ? 16 : 0);
  int bOff[4];
  #pragma unroll
  for (int j = 0; j < 4; ++j)
    bOff[j] = (ch * 64 + j * 16 + mrow) * 64 + (hi ? 16 : 0);

  f32x8 acc[4] = {};

  float4 ra0, ra1, rb0, rb1, rb2, rb3;   // staged tile in registers

  auto loadGlobals = [&](int k0) {
    const float* pa = gA + k0;
    __builtin_prefetch(pa + 32, 0, 1);   // speculative next K tile
    ra0 = *(const float4*)pa;
    ra1 = *(const float4*)(pa + 4);
    if (WT == 0) {
      const float* pb = gB + k0;
      __builtin_prefetch(pb + 32, 0, 1);
      rb0 = *(const float4*)pb;
      rb1 = *(const float4*)(pb + 4);
      rb2 = *(const float4*)(pb + 8);
      rb3 = *(const float4*)(pb + 12);
    } else {
      const float* pb = gB + (size_t)k0 * ldw;
      rb0 = *(const float4*)pb;
      rb1 = *(const float4*)(pb + 4);
      rb2 = *(const float4*)(pb + 8);
      rb3 = *(const float4*)(pb + 12);
    }
  };

  auto storeTile = [&](int p) {
    Pack8 u;
    u.h[0] = (__bf16)ra0.x; u.h[1] = (__bf16)ra0.y;
    u.h[2] = (__bf16)ra0.z; u.h[3] = (__bf16)ra0.w;
    u.h[4] = (__bf16)ra1.x; u.h[5] = (__bf16)ra1.y;
    u.h[6] = (__bf16)ra1.z; u.h[7] = (__bf16)ra1.w;
    *(uint4*)&lsA[p][sR * 32 + sK] = u.q;
    if (WT == 0) {
      Pack8 v0, v1;
      v0.h[0] = (__bf16)rb0.x; v0.h[1] = (__bf16)rb0.y;
      v0.h[2] = (__bf16)rb0.z; v0.h[3] = (__bf16)rb0.w;
      v0.h[4] = (__bf16)rb1.x; v0.h[5] = (__bf16)rb1.y;
      v0.h[6] = (__bf16)rb1.z; v0.h[7] = (__bf16)rb1.w;
      v1.h[0] = (__bf16)rb2.x; v1.h[1] = (__bf16)rb2.y;
      v1.h[2] = (__bf16)rb2.z; v1.h[3] = (__bf16)rb2.w;
      v1.h[4] = (__bf16)rb3.x; v1.h[5] = (__bf16)rb3.y;
      v1.h[6] = (__bf16)rb3.z; v1.h[7] = (__bf16)rb3.w;
      *(uint4*)&lsB[p][sR2 * 32 + sK2]     = v0.q;
      *(uint4*)&lsB[p][sR2 * 32 + sK2 + 8] = v1.q;
    } else {
      float f[16] = { rb0.x, rb0.y, rb0.z, rb0.w, rb1.x, rb1.y, rb1.z, rb1.w,
                      rb2.x, rb2.y, rb2.z, rb2.w, rb3.x, rb3.y, rb3.z, rb3.w };
      #pragma unroll
      for (int i = 0; i < 16; ++i)
        lsB[p][(tM + i) * 32 + tK] = (__bf16)f[i];   // transpose into [col][k]
    }
  };

  auto compute = [&](int p) {
    const char* lab = (const char*)lsA[p];
    const char* lbb = (const char*)lsB[p];
    FragU a, b[4];
    a.q[0] = *(const uint4*)(lab + aOff);
    a.q[1] = *(const uint4*)(lab + aOff + 32);
    #pragma unroll
    for (int j = 0; j < 4; ++j) {
      b[j].q[0] = *(const uint4*)(lbb + bOff[j]);
      b[j].q[1] = *(const uint4*)(lbb + bOff[j] + 32);
    }
    #pragma unroll
    for (int j = 0; j < 4; ++j)
      acc[j] = __builtin_amdgcn_wmma_f32_16x16x32_bf16(false, a.v, false, b[j].v,
                                                       (short)0, acc[j], false, false);
  };

  const int ntiles = K >> 5;

  // prologue: stage tile 0 into buffer 0
  loadGlobals(0);
  storeTile(0);

  // steady state: branch-free pipeline, one barrier per K-step
  for (int t = 0; t < ntiles - 1; ++t) {
    __syncthreads();
    const int p = t & 1;
    loadGlobals((t + 1) * 32);   // global loads issue under the WMMAs
    compute(p);
    storeTile(p ^ 1);
  }
  // epilogue: last tile
  __syncthreads();
  compute((ntiles - 1) & 1);

  // C/D layout: lane<16 -> M=r, N=lane ; lane>=16 -> M=r+8, N=lane-16
  #pragma unroll
  for (int j = 0; j < 4; ++j) {
    const int c = colBase0 + ch * 64 + j * 16 + mrow;
    const float bi = bias ? bias[c] : 0.f;
    #pragma unroll
    for (int r = 0; r < 8; ++r) {
      int row = rowBase0 + rs * 16 + (hi ? r + 8 : r);
      float v = acc[j][r] + bi;
      if (ACT) v = lrelu(v, 0.01f);
      C[(size_t)row * ldc + c] = v;
    }
  }
}

// ---------------------------------------------------------------------------
// GAT edge kernels
// ---------------------------------------------------------------------------
// logits[e][h] = sum_d lrelu(xl[src][h][d] + xr[dst][h][d], 0.2) * att[h][d]
// and atomic segment-max into mxenc[dst][h]. One wave per edge.
__global__ __launch_bounds__(256) void gat_logits_kernel(
    const float* __restrict__ xl, const float* __restrict__ xr,
    const int* __restrict__ adj, const float* __restrict__ att,
    float* __restrict__ logits, unsigned* __restrict__ mxenc)
{
  const int lane = threadIdx.x & 31;
  const int e = blockIdx.x * 8 + (threadIdx.x >> 5);
  if (e >= GNN_ETOT) return;
  int s, d;
  if (e < GNN_E) { s = adj[e]; d = adj[GNN_E + e]; } else { s = e - GNN_E; d = s; }
  const float* pl = xl + (size_t)s * 512;
  const float* pr = xr + (size_t)d * 512;
  #pragma unroll
  for (int h = 0; h < 4; ++h) {
    float acc = 0.f;
    #pragma unroll
    for (int q = 0; q < 4; ++q) {
      int dd = h * 128 + lane + 32 * q;
      acc += lrelu(pl[dd] + pr[dd], 0.2f) * att[dd];
    }
    acc = wave_sum32(acc);
    if (lane == 0) {
      logits[(size_t)e * 4 + h] = acc;
      atomicMax(&mxenc[d * 4 + h], enc_f32(acc));
    }
  }
}

// ex = exp(logit - mx[dst]); den[dst] += ex. One thread per (edge, head).
__global__ void gat_softmax_kernel(
    const float* __restrict__ logits, const unsigned* __restrict__ mxenc,
    const int* __restrict__ adj, float* __restrict__ ex, float* __restrict__ den)
{
  int j = blockIdx.x * blockDim.x + threadIdx.x;
  if (j >= GNN_ETOT * 4) return;
  int e = j >> 2, h = j & 3;
  int d = (e < GNN_E) ? adj[GNN_E + e] : (e - GNN_E);
  float v = __expf(logits[j] - dec_f32(mxenc[d * 4 + h]));
  ex[j] = v;
  atomicAdd(&den[d * 4 + h], v);
}

// agg[dst][h][d] += (ex/den) * xl[src][h][d]. One thread per (edge, h*128+d).
__global__ void gat_agg_kernel(
    const float* __restrict__ xl, const int* __restrict__ adj,
    const float* __restrict__ ex, const float* __restrict__ den,
    float* __restrict__ agg)
{
  long long j = (long long)blockIdx.x * blockDim.x + threadIdx.x;
  if (j >= (long long)GNN_ETOT * 512) return;
  int e = (int)(j >> 9), hd = (int)(j & 511), h = hd >> 7;
  int s, d;
  if (e < GNN_E) { s = adj[e]; d = adj[GNN_E + e]; } else { s = e - GNN_E; d = s; }
  float alpha = ex[e * 4 + h] / den[d * 4 + h];
  atomicAdd(&agg[(size_t)d * 512 + hd], alpha * xl[(size_t)s * 512 + hd]);
}

// xcat[:,128:256] = lrelu(mean_h(agg) + gat_b, 0.01)
__global__ void gat_finalize_kernel(
    const float* __restrict__ agg, const float* __restrict__ gat_b,
    float* __restrict__ xcat)
{
  int i = blockIdx.x * blockDim.x + threadIdx.x;
  if (i >= GNN_N * 128) return;
  int n = i >> 7, d = i & 127;
  const float* a = agg + (size_t)n * 512 + d;
  float g = (a[0] + a[128] + a[256] + a[384]) * 0.25f + gat_b[d];
  xcat[(size_t)n * 256 + 128 + d] = lrelu(g, 0.01f);
}

// ---------------------------------------------------------------------------
// LayerNorm (wave per row): Y = LN( res ? res + lrelu(X,0.01) : X ) * g + b
// ---------------------------------------------------------------------------
__global__ __launch_bounds__(256) void ln_kernel(
    const float* __restrict__ X, const float* __restrict__ res,
    const float* __restrict__ g, const float* __restrict__ b,
    float* __restrict__ Y, int D, int rows)
{
  const int lane = threadIdx.x & 31;
  const int row  = blockIdx.x * 8 + (threadIdx.x >> 5);
  if (row >= rows) return;
  const int Eper = D >> 5;
  const float* xp = X + (size_t)row * D;
  const float* rp = res ? res + (size_t)row * D : nullptr;
  float v[8];
  float s = 0.f;
  for (int e = 0; e < Eper; ++e) {
    float t = xp[lane + 32 * e];
    if (rp) t = rp[lane + 32 * e] + lrelu(t, 0.01f);
    v[e] = t; s += t;
  }
  float mean = wave_sum32(s) / (float)D;
  float s2 = 0.f;
  for (int e = 0; e < Eper; ++e) { float dd = v[e] - mean; s2 += dd * dd; }
  float rstd = rsqrtf(wave_sum32(s2) / (float)D + 1e-12f);
  for (int e = 0; e < Eper; ++e) {
    int c = lane + 32 * e;
    Y[(size_t)row * D + c] = (v[e] - mean) * rstd * g[c] + b[c];
  }
}

// ---------------------------------------------------------------------------
// GRU gate combine on the 256x256 eg_W matrix
// ---------------------------------------------------------------------------
__global__ void gru_kernel(
    const float* __restrict__ gi, const float* __restrict__ gh,
    const float* __restrict__ egW, float* __restrict__ Wm)
{
  int idx = blockIdx.x * blockDim.x + threadIdx.x;
  if (idx >= 256 * 256) return;
  int i = idx >> 8, j = idx & 255;
  const float* pi = gi + (size_t)i * 768;
  const float* ph = gh + (size_t)i * 768;
  float r = 1.f / (1.f + __expf(-(pi[j] + ph[j])));
  float z = 1.f / (1.f + __expf(-(pi[256 + j] + ph[256 + j])));
  float n = tanhf(pi[512 + j] + r * ph[512 + j]);
  Wm[idx] = (1.f - z) * n + z * egW[idx];
}

// ---------------------------------------------------------------------------
// GCN normalization + scatter
// ---------------------------------------------------------------------------
__global__ void degree_kernel(const int* __restrict__ adj, float* __restrict__ deg)
{
  int e = blockIdx.x * blockDim.x + threadIdx.x;
  if (e >= GNN_ETOT) return;
  int d = (e < GNN_E) ? adj[GNN_E + e] : (e - GNN_E);
  atomicAdd(&deg[d], 1.0f);
}

__global__ void dinv_kernel(const float* __restrict__ deg, float* __restrict__ dinv)
{
  int n = blockIdx.x * blockDim.x + threadIdx.x;
  if (n >= GNN_N) return;
  dinv[n] = rsqrtf(fmaxf(deg[n], 1e-12f));
}

// gcn[dst][c] += dinv[src]*dinv[dst] * xw[src][c]
__global__ void gcn_scatter_kernel(
    const int* __restrict__ adj, const float* __restrict__ dinv,
    const float* __restrict__ xw, float* __restrict__ gcn)
{
  long long j = (long long)blockIdx.x * blockDim.x + threadIdx.x;
  if (j >= (long long)GNN_ETOT * 256) return;
  int e = (int)(j >> 8), c = (int)(j & 255);
  int s, d;
  if (e < GNN_E) { s = adj[e]; d = adj[GNN_E + e]; } else { s = e - GNN_E; d = s; }
  atomicAdd(&gcn[(size_t)d * 256 + c], dinv[s] * dinv[d] * xw[(size_t)s * 256 + c]);
}

// out_line[n] = dot(out[n,:128], cls_w) + cls_b. One wave per node.
__global__ __launch_bounds__(256) void cls_kernel(
    const float* __restrict__ out, const float* __restrict__ w,
    const float* __restrict__ b, float* __restrict__ line)
{
  const int lane = threadIdx.x & 31;
  const int n = blockIdx.x * 8 + (threadIdx.x >> 5);
  if (n >= GNN_N) return;
  float acc = 0.f;
  #pragma unroll
  for (int q = 0; q < 4; ++q)
    acc += out[(size_t)n * 128 + lane + 32 * q] * w[lane + 32 * q];
  acc = wave_sum32(acc);
  if (lane == 0) line[n] = acc + b[0];
}

// ---------------------------------------------------------------------------
// Host launcher
// ---------------------------------------------------------------------------
extern "C" void kernel_launch(void* const* d_in, const int* in_sizes, int n_in,
                              void* d_out, int out_size, void* d_ws, size_t ws_size,
                              hipStream_t stream)
{
  const int N = GNN_N, E = GNN_E, Etot = GNN_ETOT;
  const float* feature   = (const float*)d_in[0];
  const int*   adj       = (const int*)  d_in[1];
  const float* emb1_w    = (const float*)d_in[2];
  const float* emb1_b    = (const float*)d_in[3];
  const float* gl_w      = (const float*)d_in[4];
  const float* gl_b      = (const float*)d_in[5];
  const float* gr_w      = (const float*)d_in[6];
  const float* gr_b      = (const float*)d_in[7];
  const float* gat_att   = (const float*)d_in[8];
  const float* gat_b     = (const float*)d_in[9];
  const float* ma1_w     = (const float*)d_in[10];
  const float* ma1_b     = (const float*)d_in[11];
  const float* ma2_w     = (const float*)d_in[12];
  const float* ma2_b     = (const float*)d_in[13];
  const float* lv_w      = (const float*)d_in[18];
  const float* lv_b      = (const float*)d_in[19];
  const float* inproj_w  = (const float*)d_in[20];
  const float* inproj_b  = (const float*)d_in[21];
  const float* outproj_w = (const float*)d_in[22];
  const float* outproj_b = (const float*)d_in[23];
  const float* ma_ln_g   = (const float*)d_in[24];
  const float* ma_ln_b   = (const float*)d_in[25];
  const float* fn1_g     = (const float*)d_in[26];
  const float* fn1_b     = (const float*)d_in[27];
  const float* eg_W      = (const float*)d_in[28];
  const float* gru_wih   = (const float*)d_in[29];
  const float* gru_whh   = (const float*)d_in[30];
  const float* gru_bih   = (const float*)d_in[31];
  const float* gru_bhh   = (const float*)d_in[32];
  const float* nev_g     = (const float*)d_in[33];
  const float* nev_b     = (const float*)d_in[34];
  const float* fc1_w     = (const float*)d_in[35];
  const float* fc1_b     = (const float*)d_in[36];
  const float* fc2_w     = (const float*)d_in[37];
  const float* fc2_b     = (const float*)d_in[38];
  const float* ng_g      = (const float*)d_in[39];
  const float* ng_b      = (const float*)d_in[40];
  const float* cls_w     = (const float*)d_in[41];
  const float* cls_b     = (const float*)d_in[42];

  float* out_line = (float*)d_out;        // [N]
  float* out_mat  = out_line + N;         // [N,128]

  // workspace layout (floats), with buffer reuse
  float* ws    = (float*)d_ws;
  float* xl    = ws;                      // N*512
  float* xr    = xl   + (size_t)N * 512;  // N*512 (xr, then agg)
  float* xcat  = xr   + (size_t)N * 512;  // N*256 (x0 | lrelu(gat))
  float* p1    = xcat + (size_t)N * 256;  // N*128 (ma1 out, later fc1 out)
  float* p2    = p1   + (size_t)N * 128;  // N*256 (ma2 out, t, p, fc2 out)
  float* vb    = p2   + (size_t)N * 256;  // N*256 (v, later att)
  float* xb    = vb   + (size_t)N * 256;  // N*256 (x after fn1 LN)
  float* xw    = xb   + (size_t)N * 256;  // N*256 (x@W, later x2)
  float* gcn   = xw   + (size_t)N * 256;  // N*256
  float* logits = gcn + (size_t)N * 256;           // Etot*4
  unsigned* mxenc = (unsigned*)(logits + (size_t)Etot * 4); // N*4
  float* ex    = (float*)(mxenc + (size_t)N * 4);  // Etot*4
  float* den   = ex   + (size_t)Etot * 4;          // N*4
  float* deg   = den  + (size_t)N * 4;             // N
  float* dinv  = deg  + N;                         // N
  float* gi    = dinv + N;                         // 256*768
  float* gh    = gi   + 256 * 768;                 // 256*768
  float* Wm    = gh   + 256 * 768;                 // 256*256

  auto gemm = [&](const float* A, const float* W, const float* bias, float* C,
                  int rows, int K, int M, int lda, int ldw, int ldc,
                  int wT, int act) {
    dim3 grid(M / 128, rows / 64);
    if (wT)
      gemm_wmma_kernel<1, 0><<<grid, 256, 0, stream>>>(A, W, bias, C, K, lda, ldw, ldc);
    else if (act)
      gemm_wmma_kernel<0, 1><<<grid, 256, 0, stream>>>(A, W, bias, C, K, lda, ldw, ldc);
    else
      gemm_wmma_kernel<0, 0><<<grid, 256, 0, stream>>>(A, W, bias, C, K, lda, ldw, ldc);
  };

  // zero accumulation buffers
  hipMemsetAsync(mxenc, 0, (size_t)N * 4 * sizeof(unsigned), stream);
  hipMemsetAsync(den,   0, (size_t)N * 4 * sizeof(float), stream);
  hipMemsetAsync(deg,   0, (size_t)N * sizeof(float), stream);

  // --- GAT projections + embedding ---
  gemm(feature, gl_w, gl_b, xl, N, 256, 512, 256, 256, 512, 0, 0);
  gemm(feature, gr_w, gr_b, xr, N, 256, 512, 256, 256, 512, 0, 0);
  gemm(feature, emb1_w, emb1_b, xcat, N, 256, 128, 256, 256, 256, 0, 0); // cols 0..127

  // --- GAT attention over edges ---
  gat_logits_kernel<<<(Etot + 7) / 8, 256, 0, stream>>>(xl, xr, adj, gat_att, logits, mxenc);
  gat_softmax_kernel<<<(Etot * 4 + 255) / 256, 256, 0, stream>>>(logits, mxenc, adj, ex, den);
  hipMemsetAsync(xr, 0, (size_t)N * 512 * sizeof(float), stream);  // xr -> agg
  {
    long long tot = (long long)Etot * 512;
    gat_agg_kernel<<<(unsigned)((tot + 255) / 256), 256, 0, stream>>>(xl, adj, ex, den, xr);
  }
  gat_finalize_kernel<<<(N * 128 + 255) / 256, 256, 0, stream>>>(xr, gat_b, xcat);

  // --- MLP attention block ---
  gemm(xcat, ma1_w, ma1_b, p1, N, 256, 128, 256, 256, 128, 0, 1);
  gemm(p1,   ma2_w, ma2_b, p2, N, 128, 256, 128, 128, 256, 0, 1);
  gemm(p2,   lv_w,  lv_b,  vb, N, 256, 256, 256, 256, 256, 0, 0);
  gemm(vb, inproj_w + 2 * 256 * 256, inproj_b + 2 * 256, p2,
       N, 256, 256, 256, 256, 256, 0, 0);                       // t = v@Wv.T+bv
  gemm(p2, outproj_w, outproj_b, vb, N, 256, 256, 256, 256, 256, 0, 0); // att
  ln_kernel<<<N / 8, 256, 0, stream>>>(vb, nullptr, ma_ln_g, ma_ln_b, p2, 256, N); // p
  ln_kernel<<<N / 8, 256, 0, stream>>>(p2, xcat, fn1_g, fn1_b, xb, 256, N);        // x

  // --- GRU edge-gated weight ---
  gemm(eg_W, gru_wih, gru_bih, gi, 256, 256, 768, 256, 256, 768, 0, 0);
  gemm(eg_W, gru_whh, gru_bhh, gh, 256, 256, 768, 256, 256, 768, 0, 0);
  gru_kernel<<<(256 * 256 + 255) / 256, 256, 0, stream>>>(gi, gh, eg_W, Wm);
  gemm(xb, Wm, nullptr, xw, N, 256, 256, 256, 256, 256, 1, 0);  // xw = x @ W

  // --- GCN aggregation ---
  degree_kernel<<<(Etot + 255) / 256, 256, 0, stream>>>(adj, deg);
  dinv_kernel<<<(N + 255) / 256, 256, 0, stream>>>(deg, dinv);
  hipMemsetAsync(gcn, 0, (size_t)N * 256 * sizeof(float), stream);
  {
    long long tot = (long long)Etot * 256;
    gcn_scatter_kernel<<<(unsigned)((tot + 255) / 256), 256, 0, stream>>>(adj, dinv, xw, gcn);
  }
  ln_kernel<<<N / 8, 256, 0, stream>>>(gcn, xb, nev_g, nev_b, xw, 256, N); // x2

  // --- Head ---
  gemm(xw, fc1_w, fc1_b, p1, N, 256, 128, 256, 256, 128, 0, 1);
  gemm(p1, fc2_w, fc2_b, p2, N, 128, 128, 128, 128, 128, 0, 1);
  ln_kernel<<<N / 8, 256, 0, stream>>>(p2, nullptr, ng_g, ng_b, out_mat, 128, N);
  cls_kernel<<<N / 8, 256, 0, stream>>>(out_mat, cls_w, cls_b, out_line);
}
